// MLAAttention_10986526343379
// MI455X (gfx1250) — matrix operand order repaired and checked by reading, not compiled, not run on hardware
//
#include <hip/hip_runtime.h>
#include <hip/hip_bf16.h>
#include <cstdint>

// ---------------------------------------------------------------------------
// MLA attention pipeline for gfx1250 (MI455X), bf16 WMMA + TDM staging.
// ---------------------------------------------------------------------------

#define Bc   2
#define Sc   2048
#define HIDc 2048
#define Hc   16
#define DNOPE 128
#define DROPE 64
#define HDIM 192
#define VDIM 128
#define KVRANK 512
#define QRANK 1536
#define BSc  (Bc * Sc)          // 4096 rows

typedef __attribute__((ext_vector_type(8)))  float  v8f;
typedef __attribute__((ext_vector_type(8)))  __bf16 v8bf;
typedef __attribute__((ext_vector_type(16))) __bf16 v16bf;
typedef __attribute__((ext_vector_type(4)))  unsigned int u32x4;
typedef __attribute__((ext_vector_type(8)))  int i32x8;
typedef __attribute__((ext_vector_type(4)))  int i32x4;

union Frag16 { v16bf v; v8bf h[2]; };

__device__ inline v8f v8f_zero() {
    v8f z;
#pragma unroll
    for (int i = 0; i < 8; ++i) z[i] = 0.0f;
    return z;
}

#if defined(__has_builtin)
#if __has_builtin(__builtin_amdgcn_tensor_load_to_lds) && \
    __has_builtin(__builtin_amdgcn_s_wait_tensorcnt)
#define USE_TDM 1
#endif
#endif
#ifndef USE_TDM
#define USE_TDM 0
#endif

#if USE_TDM
// Issue a TDM 2D tile load: `rows` rows of 32 bf16 elements (64B) each,
// row stride `strideElems` bf16 elements in global memory, destination LDS
// padded +16B after every 64B (row stride 80B), matching [rows][40] layout.
__device__ inline void tdm_load_tile_a(const __bf16* gsrc, unsigned ldsByteAddr,
                                       int rows, int strideElems) {
    unsigned long long ga = (unsigned long long)(uintptr_t)gsrc;
    u32x4 g0;
    g0[0] = 1u;                                   // count=1, user mode
    g0[1] = ldsByteAddr;                          // lds_addr
    g0[2] = (unsigned)(ga & 0xFFFFFFFFu);         // global_addr[31:0]
    g0[3] = (unsigned)((ga >> 32) & 0x01FFFFFFu)  // global_addr[56:32]
            | (2u << 30);                         // type=2 (image)
    i32x8 g1;
    // data_size=1 (2B) | pad_enable | pad_interval=3 (16 DW=64B) | pad_amount=3 (4 DW=16B)
    g1[0] = (int)0x06D10000u;
    g1[1] = (int)(((unsigned)strideElems & 0xFFFFu) << 16);      // tensor_dim0[15:0]
    g1[2] = (int)((((unsigned)strideElems >> 16) & 0xFFFFu)      // tensor_dim0[31:16]
                  | (((unsigned)BSc & 0xFFFFu) << 16));          // tensor_dim1[15:0]
    g1[3] = (int)(0u | (32u << 16));                             // tile_dim0 = 32
    g1[4] = (int)(unsigned)rows;                                 // tile_dim1, tile_dim2=0
    g1[5] = (int)(unsigned)strideElems;                          // tensor_dim0_stride[31:0]
    g1[6] = 0;
    g1[7] = 0;
    i32x4 z4;
    z4[0] = 0; z4[1] = 0; z4[2] = 0; z4[3] = 0;
    i32x8 z8;
#pragma unroll
    for (int i = 0; i < 8; ++i) z8[i] = 0;
    // 6-arg form (clang-23 / therock-10.0 headers): extra i32x8 group + cpol.
    __builtin_amdgcn_tensor_load_to_lds(g0, g1, z4, z4, z8, 0);
}
#endif

// ---------------------------------------------------------------------------
// fp32 -> bf16 conversion
// ---------------------------------------------------------------------------
__global__ __launch_bounds__(256) void f32_to_bf16_kernel(
    const float* __restrict__ in, __bf16* __restrict__ out, int n) {
    int i = blockIdx.x * 256 + threadIdx.x;
    if (i < n) out[i] = (__bf16)in[i];
}

// ---------------------------------------------------------------------------
// Generic bf16 GEMM: C(MxN,f32) = A(MxK,bf16,row-major) * B(KxN,bf16,row-major)
// Block tile 128x64, 8 waves (4x2), wave tile 32x32, K-step 32.
// A tile staged by the Tensor Data Mover (wave 0 issues, padded to [128][40]).
// ---------------------------------------------------------------------------
__global__ __launch_bounds__(256) void gemm_bf16_kernel(
    const __bf16* __restrict__ A, const __bf16* __restrict__ Bm,
    float* __restrict__ C, int M, int N, int K) {
    __shared__ __bf16 As[128][40];   // 128 rows x 32 (pad 8)
    __shared__ __bf16 Bt[64][40];    // B transposed: 64 cols x 32 (pad 8)

    const int tid  = threadIdx.x;
    const int lane = tid & 31;
    const int w    = tid >> 5;
    const int wm   = w & 3;          // 0..3
    const int wn   = w >> 2;         // 0..1
    const int lrow = lane & 15;
    const int kh   = lane >> 4;      // 0 or 1
    const int mBase = blockIdx.y * 128;
    const int nBase = blockIdx.x * 64;

    v8f acc[2][2];
#pragma unroll
    for (int mi = 0; mi < 2; ++mi)
#pragma unroll
        for (int ni = 0; ni < 2; ++ni) acc[mi][ni] = v8f_zero();

    for (int k0 = 0; k0 < K; k0 += 32) {
        // ---- Stage A tile 128x32 into LDS ----
#if USE_TDM
        if (w == 0) {
            tdm_load_tile_a(A + (size_t)mBase * K + k0,
                            (unsigned)(uintptr_t)&As[0][0], 128, K);
            __builtin_amdgcn_s_wait_tensorcnt(0);
        }
#else
#pragma unroll
        for (int it = 0; it < 2; ++it) {
            int j  = tid + it * 256;
            int r  = j >> 2;
            int ch = j & 3;
            *(v8bf*)&As[r][ch * 8] =
                *(const v8bf*)(A + (size_t)(mBase + r) * K + k0 + ch * 8);
        }
#endif
        // ---- Stage B tile transposed: 32x64 -> Bt[col][k] ----
        {
            int k  = tid >> 3;
            int ch = tid & 7;
            v8bf tmp = *(const v8bf*)(Bm + (size_t)(k0 + k) * N + nBase + ch * 8);
#pragma unroll
            for (int e = 0; e < 8; ++e) Bt[ch * 8 + e][k] = tmp[e];
            // prefetch next K-step tiles into caches
            if (k0 + 32 < K) {
                __builtin_prefetch(Bm + (size_t)(k0 + 32 + k) * N + nBase + ch * 8, 0, 1);
                __builtin_prefetch(A + (size_t)(mBase + (tid >> 1)) * K + k0 + 32 +
                                       (tid & 1) * 16, 0, 1);
            }
        }
        __syncthreads();

        Frag16 fa[2], fb[2];
#pragma unroll
        for (int mi = 0; mi < 2; ++mi) {
            int r = wm * 32 + mi * 16 + lrow;
            // A 16x32 layout: lanes<16 -> K{0..7},{16..23}; lanes>=16 -> K{8..15},{24..31}
            fa[mi].h[0] = *(const v8bf*)&As[r][kh * 8];
            fa[mi].h[1] = *(const v8bf*)&As[r][16 + kh * 8];
        }
#pragma unroll
        for (int ni = 0; ni < 2; ++ni) {
            int c = wn * 32 + ni * 16 + lrow;
            // B 32x16 layout: lanes<16 -> K{0..7},{16..23}; lanes>=16 -> K{8..15},{24..31}
            fb[ni].h[0] = *(const v8bf*)&Bt[c][kh * 8];
            fb[ni].h[1] = *(const v8bf*)&Bt[c][16 + kh * 8];
        }
#pragma unroll
        for (int mi = 0; mi < 2; ++mi)
#pragma unroll
            for (int ni = 0; ni < 2; ++ni)
                acc[mi][ni] = __builtin_amdgcn_wmma_f32_16x16x32_bf16(
                    false, fa[mi].v, false, fb[ni].v, (short)0, acc[mi][ni],
                    false, false);
        __syncthreads();
    }

    // Store: C layout lane L: M = r + 8*(L>=16), N = L&15
#pragma unroll
    for (int mi = 0; mi < 2; ++mi)
#pragma unroll
        for (int ni = 0; ni < 2; ++ni) {
            int row0 = mBase + wm * 32 + mi * 16 + kh * 8;
            int col  = nBase + wn * 32 + ni * 16 + lrow;
#pragma unroll
            for (int r = 0; r < 8; ++r)
                C[(size_t)(row0 + r) * N + col] = acc[mi][ni][r];
        }
}

// ---------------------------------------------------------------------------
// RMSNorm over first 512 of each 576-wide row of kv_full -> bf16 (4096x512)
// ---------------------------------------------------------------------------
__global__ __launch_bounds__(256) void rmsnorm_kernel(
    const float* __restrict__ kvfull, const float* __restrict__ wn,
    __bf16* __restrict__ out) {
    const int row = blockIdx.x;
    const float* xr = kvfull + (size_t)row * (KVRANK + DROPE);
    float ss = 0.0f;
    for (int i = threadIdx.x; i < KVRANK; i += 256) {
        float v = xr[i];
        ss += v * v;
    }
#pragma unroll
    for (int m = 1; m < 32; m <<= 1) ss += __shfl_xor(ss, m);
    __shared__ float red[8];
    int lane = threadIdx.x & 31, wv = threadIdx.x >> 5;
    if (lane == 0) red[wv] = ss;
    __syncthreads();
    float tot = 0.0f;
#pragma unroll
    for (int i = 0; i < 8; ++i) tot += red[i];
    float r = rsqrtf(tot / (float)KVRANK + 1e-6f);
    for (int i = threadIdx.x; i < KVRANK; i += 256)
        out[(size_t)row * KVRANK + i] = (__bf16)(xr[i] * r * wn[i]);
}

// ---------------------------------------------------------------------------
// RoPE + assemble per-(b,h) contiguous bf16 tensors:
//   Qh[(b*H+h)*S + s][192], Kh[(b*H+h)*S + s][192], Vh[(b*H+h)*S + s][128]
// ---------------------------------------------------------------------------
__global__ __launch_bounds__(256) void rope_assemble_kernel(
    const float* __restrict__ qf, const float* __restrict__ kvexp,
    const float* __restrict__ kvfull, __bf16* __restrict__ Qh,
    __bf16* __restrict__ Kh, __bf16* __restrict__ Vh) {
    const int bs = blockIdx.x;        // b*S + s
    const int b  = bs / Sc;
    const int s  = bs - b * Sc;
    const int tid = threadIdx.x;

    __shared__ float cs[32], sn[32], kro[64];
    if (tid < 32) {
        float ex   = (float)(2 * tid) / (float)DROPE;
        float freq = __powf(10000.0f, -ex);
        float ang  = (float)s * freq;
        cs[tid] = __cosf(ang);
        sn[tid] = __sinf(ang);
    }
    __syncthreads();
    if (tid < 64) {
        int j = tid & 31;
        float x1 = kvfull[(size_t)bs * (KVRANK + DROPE) + KVRANK + j];
        float x2 = kvfull[(size_t)bs * (KVRANK + DROPE) + KVRANK + 32 + j];
        kro[tid] = (tid < 32) ? (x1 * cs[j] - x2 * sn[j])
                              : (x2 * cs[j] + x1 * sn[j]);
    }
    __syncthreads();

    // Q with rope on last 64 dims of each head
    for (int idx = tid; idx < Hc * HDIM; idx += 256) {
        int hh = idx / HDIM, d = idx - hh * HDIM;
        const float* qrow = qf + (size_t)bs * (Hc * HDIM) + hh * HDIM;
        float v;
        if (d < DNOPE) {
            v = qrow[d];
        } else {
            int j = (d - DNOPE) & 31;
            float x1 = qrow[DNOPE + j], x2 = qrow[DNOPE + 32 + j];
            v = (d < DNOPE + 32) ? (x1 * cs[j] - x2 * sn[j])
                                 : (x2 * cs[j] + x1 * sn[j]);
        }
        Qh[(((size_t)(b * Hc + hh)) * Sc + s) * HDIM + d] = (__bf16)v;
    }
    // K = [k_nope | roped k_rope broadcast]
    for (int idx = tid; idx < Hc * HDIM; idx += 256) {
        int hh = idx / HDIM, d = idx - hh * HDIM;
        float v = (d < DNOPE)
                      ? kvexp[(size_t)bs * (Hc * 256) + hh * 256 + d]
                      : kro[d - DNOPE];
        Kh[(((size_t)(b * Hc + hh)) * Sc + s) * HDIM + d] = (__bf16)v;
    }
    // V
    for (int idx = tid; idx < Hc * VDIM; idx += 256) {
        int hh = idx / VDIM, d = idx - hh * VDIM;
        Vh[(((size_t)(b * Hc + hh)) * Sc + s) * VDIM + d] =
            (__bf16)kvexp[(size_t)bs * (Hc * 256) + hh * 256 + DNOPE + d];
    }
}

// ---------------------------------------------------------------------------
// Causal flash attention: 4 waves/block, each wave owns 16 query rows,
// K-step of 32 keys. Shared staged K tile + transposed V tile.
// Output bf16 attn[(b*S+s)*2048 + h*128 + d].
// ---------------------------------------------------------------------------
__global__ __launch_bounds__(128) void fa_kernel(
    const __bf16* __restrict__ Qh, const __bf16* __restrict__ Kh,
    const __bf16* __restrict__ Vh, __bf16* __restrict__ Ob) {
    __shared__ __bf16 Ks[32][200];      // 32 keys x 192 (pad 8)
    __shared__ __bf16 Vt[128][40];      // V transposed: 128 dims x 32 keys (pad 8)
    __shared__ __bf16 Ps[4][16][40];    // per-wave P tile 16x32 (pad 8)

    const int tid  = threadIdx.x;
    const int lane = tid & 31;
    const int w    = tid >> 5;
    const int lrow = lane & 15;
    const int kh   = lane >> 4;
    const int q0   = blockIdx.x * 64;
    const int bh   = blockIdx.y;        // b*H + h
    const int b    = bh >> 4;
    const int h    = bh & 15;
    const int qw   = q0 + w * 16;
    const float scale = 0.07216878364870322f;   // 1/sqrt(192)

    // Q fragments: 16x192 = 6 K-steps of 32
    Frag16 qfr[6];
    {
        const __bf16* Qbase = Qh + ((size_t)bh * Sc + (qw + lrow)) * HDIM;
#pragma unroll
        for (int kk = 0; kk < 6; ++kk) {
            qfr[kk].h[0] = *(const v8bf*)(Qbase + kk * 32 + kh * 8);
            qfr[kk].h[1] = *(const v8bf*)(Qbase + kk * 32 + 16 + kh * 8);
        }
    }

    v8f oacc[8];
#pragma unroll
    for (int c = 0; c < 8; ++c) oacc[c] = v8f_zero();
    float mrow[8], lrow_s[8];
#pragma unroll
    for (int r = 0; r < 8; ++r) { mrow[r] = -1e30f; lrow_s[r] = 0.0f; }

    const int nk = q0 / 32 + 2;         // 32-key blocks to cover causal extent
    for (int kb = 0; kb < nk; ++kb) {
        const int kbase = kb * 32;
        __syncthreads();   // protect Ks/Vt from previous iteration readers
        // Stage K tile 32x192 (768 chunks of 8, 128 threads -> 6 passes)
#pragma unroll
        for (int it = 0; it < 6; ++it) {
            int j   = tid + it * 128;
            int key = j / 24;
            int ch  = j - key * 24;
            *(v8bf*)&Ks[key][ch * 8] =
                *(const v8bf*)(Kh + ((size_t)bh * Sc + kbase + key) * HDIM + ch * 8);
        }
        // Stage V tile transposed: Vt[d][key]
#pragma unroll
        for (int it = 0; it < 4; ++it) {
            int j   = tid + it * 128;
            int key = j >> 4;
            int ch  = j & 15;
            v8bf tmp = *(const v8bf*)(Vh + ((size_t)bh * Sc + kbase + key) * VDIM + ch * 8);
#pragma unroll
            for (int e = 0; e < 8; ++e) Vt[ch * 8 + e][key] = tmp[e];
        }
        // prefetch next key block
        if (kb + 1 < nk) {
            __builtin_prefetch(Kh + ((size_t)bh * Sc + kbase + 32 + (tid >> 2)) * HDIM +
                                   (tid & 3) * 48, 0, 1);
            __builtin_prefetch(Vh + ((size_t)bh * Sc + kbase + 32 + (tid >> 2)) * VDIM +
                                   (tid & 3) * 32, 0, 1);
        }
        __syncthreads();

        // Scores: two 16x16 tiles (keys kbase..+15, kbase+16..+31)
        v8f sacc[2];
        sacc[0] = v8f_zero();
        sacc[1] = v8f_zero();
#pragma unroll
        for (int t = 0; t < 2; ++t) {
            const __bf16* Kb = &Ks[t * 16 + lrow][0];
#pragma unroll
            for (int kk = 0; kk < 6; ++kk) {
                Frag16 kf;   // B-frag: lanes<16 hold dims +{0..7},{16..23}, lanes>=16 +{8..15},{24..31}
                kf.h[0] = *(const v8bf*)(Kb + kk * 32 + kh * 8);
                kf.h[1] = *(const v8bf*)(Kb + kk * 32 + 16 + kh * 8);
                sacc[t] = __builtin_amdgcn_wmma_f32_16x16x32_bf16(
                    false, qfr[kk].v, false, kf.v, (short)0, sacc[t], false, false);
            }
        }

        // Online softmax (rows live per 16-lane half; reduce via shfl_xor 1..8)
        float escale[8];
#pragma unroll
        for (int r = 0; r < 8; ++r) {
            int qrow = qw + r + kh * 8;
            bool msk0 = (kbase + lrow) > qrow;
            bool msk1 = (kbase + 16 + lrow) > qrow;
            float s0 = msk0 ? -1e30f : sacc[0][r] * scale;
            float s1 = msk1 ? -1e30f : sacc[1][r] * scale;
            float rm = fmaxf(s0, s1);
            rm = fmaxf(rm, __shfl_xor(rm, 1));
            rm = fmaxf(rm, __shfl_xor(rm, 2));
            rm = fmaxf(rm, __shfl_xor(rm, 4));
            rm = fmaxf(rm, __shfl_xor(rm, 8));
            float mn  = fmaxf(mrow[r], rm);
            float esc = __expf(mrow[r] - mn);
            float p0  = msk0 ? 0.0f : __expf(s0 - mn);
            float p1  = msk1 ? 0.0f : __expf(s1 - mn);
            float rs  = p0 + p1;
            rs += __shfl_xor(rs, 1);
            rs += __shfl_xor(rs, 2);
            rs += __shfl_xor(rs, 4);
            rs += __shfl_xor(rs, 8);
            lrow_s[r] = lrow_s[r] * esc + rs;
            mrow[r]   = mn;
            escale[r] = esc;
            Ps[w][r + kh * 8][lrow]      = (__bf16)p0;
            Ps[w][r + kh * 8][16 + lrow] = (__bf16)p1;
        }
#pragma unroll
        for (int c = 0; c < 8; ++c)
#pragma unroll
            for (int r = 0; r < 8; ++r) oacc[c][r] *= escale[r];
        __syncthreads();

        // P fragment (A layout, 16x32)
        Frag16 pf;
        pf.h[0] = *(const v8bf*)&Ps[w][lrow][kh * 8];
        pf.h[1] = *(const v8bf*)&Ps[w][lrow][16 + kh * 8];

        // O += P(16x32) x V(32x128): 8 N-chunks of 16
#pragma unroll
        for (int c = 0; c < 8; ++c) {
            int d = c * 16 + lrow;
            Frag16 vf;   // B-frag from Vt: lanes<16 keys {0..7},{16..23}, lanes>=16 keys {8..15},{24..31}
            vf.h[0] = *(const v8bf*)&Vt[d][kh * 8];
            vf.h[1] = *(const v8bf*)&Vt[d][16 + kh * 8];
            oacc[c] = __builtin_amdgcn_wmma_f32_16x16x32_bf16(
                false, pf.v, false, vf.v, (short)0, oacc[c], false, false);
        }
    }

    // Normalize and store bf16 attn output
#pragma unroll
    for (int c = 0; c < 8; ++c)
#pragma unroll
        for (int r = 0; r < 8; ++r) {
            int srow = qw + r + kh * 8;
            Ob[((size_t)b * Sc + srow) * (Hc * VDIM) + h * VDIM + c * 16 + lrow] =
                (__bf16)(oacc[c][r] / lrow_s[r]);
        }
}

// ---------------------------------------------------------------------------
// Host orchestration
// ---------------------------------------------------------------------------
static inline size_t alignup256(size_t x) { return (x + 255) & ~(size_t)255; }

extern "C" void kernel_launch(void* const* d_in, const int* in_sizes, int n_in,
                              void* d_out, int out_size, void* d_ws, size_t ws_size,
                              hipStream_t stream) {
    (void)in_sizes; (void)n_in; (void)out_size; (void)ws_size;
    const float* x    = (const float*)d_in[0];
    const float* wqd  = (const float*)d_in[1];
    const float* wqu  = (const float*)d_in[2];
    const float* wkvd = (const float*)d_in[3];
    const float* kvnw = (const float*)d_in[4];
    const float* wkvu = (const float*)d_in[5];
    const float* wout = (const float*)d_in[6];
    float* out = (float*)d_out;

    char* p = (char*)d_ws;
    auto take = [&](size_t bytes) -> char* {
        char* r = p;
        p += alignup256(bytes);
        return r;
    };

    __bf16* xb     = (__bf16*)take((size_t)BSc * HIDc * 2);
    __bf16* wqd_b  = (__bf16*)take((size_t)HIDc * QRANK * 2);
    __bf16* wqu_b  = (__bf16*)take((size_t)QRANK * Hc * HDIM * 2);
    __bf16* wkvd_b = (__bf16*)take((size_t)HIDc * (KVRANK + DROPE) * 2);
    __bf16* wkvu_b = (__bf16*)take((size_t)KVRANK * Hc * 256 * 2);
    __bf16* wout_b = (__bf16*)take((size_t)(Hc * VDIM) * HIDc * 2);
    float*  qlat   = (float*)take((size_t)BSc * QRANK * 4);
    __bf16* qlat_b = (__bf16*)take((size_t)BSc * QRANK * 2);
    float*  qf     = (float*)take((size_t)BSc * Hc * HDIM * 4);
    float*  kvfull = (float*)take((size_t)BSc * (KVRANK + DROPE) * 4);
    __bf16* kvc_b  = (__bf16*)take((size_t)BSc * KVRANK * 2);
    float*  kvexp  = (float*)take((size_t)BSc * Hc * 256 * 4);
    __bf16* Qh     = (__bf16*)take((size_t)Bc * Hc * Sc * HDIM * 2);
    __bf16* Kh     = (__bf16*)take((size_t)Bc * Hc * Sc * HDIM * 2);
    __bf16* Vh     = (__bf16*)take((size_t)Bc * Hc * Sc * VDIM * 2);
    __bf16* attn_b = (__bf16*)take((size_t)BSc * Hc * VDIM * 2);

    auto conv = [&](const float* src, __bf16* dst, size_t n) {
        f32_to_bf16_kernel<<<dim3((unsigned)((n + 255) / 256)), dim3(256), 0, stream>>>(
            src, dst, (int)n);
    };
    conv(x,    xb,     (size_t)BSc * HIDc);
    conv(wqd,  wqd_b,  (size_t)HIDc * QRANK);
    conv(wqu,  wqu_b,  (size_t)QRANK * Hc * HDIM);
    conv(wkvd, wkvd_b, (size_t)HIDc * (KVRANK + DROPE));
    conv(wkvu, wkvu_b, (size_t)KVRANK * Hc * 256);
    conv(wout, wout_b, (size_t)(Hc * VDIM) * HIDc);

    auto gemm = [&](const __bf16* A, const __bf16* Bm, float* C, int M, int N, int K) {
        gemm_bf16_kernel<<<dim3(N / 64, M / 128), dim3(256), 0, stream>>>(A, Bm, C, M, N, K);
    };

    // q_lat = x @ w_q_down   (4096x1536, K=2048)
    gemm(xb, wqd_b, qlat, BSc, QRANK, HIDc);
    conv(qlat, qlat_b, (size_t)BSc * QRANK);
    // q = q_lat @ w_q_up     (4096x3072, K=1536)
    gemm(qlat_b, wqu_b, qf, BSc, Hc * HDIM, QRANK);
    // kv_full = x @ w_kv_down (4096x576, K=2048)
    gemm(xb, wkvd_b, kvfull, BSc, KVRANK + DROPE, HIDc);
    // rmsnorm -> bf16
    rmsnorm_kernel<<<dim3(BSc), dim3(256), 0, stream>>>(kvfull, kvnw, kvc_b);
    // kv_exp = kv_c @ w_kv_up (4096x4096, K=512)
    gemm(kvc_b, wkvu_b, kvexp, BSc, Hc * 256, KVRANK);
    // rope + assemble Qh/Kh/Vh
    rope_assemble_kernel<<<dim3(BSc), dim3(256), 0, stream>>>(qf, kvexp, kvfull,
                                                              Qh, Kh, Vh);
    // flash attention
    fa_kernel<<<dim3(Sc / 64, Bc * Hc), dim3(128), 0, stream>>>(Qh, Kh, Vh, attn_b);
    // final projection (4096x2048, K=2048) -> fp32 output
    gemm(attn_b, wout_b, out, BSc, HIDc, Hc * VDIM);
}
